// MainNetwork_40441412059856
// MI455X (gfx1250) — matrix operand order repaired and checked by reading, not compiled, not run on hardware
//
#include <hip/hip_runtime.h>

typedef __attribute__((ext_vector_type(16))) _Float16 v16h;
typedef __attribute__((ext_vector_type(8)))  _Float16 v8h;
typedef __attribute__((ext_vector_type(8)))  float    v8f;

#define N_SAMP 512
#define R_ROOM 64
#define WMR    6
#define HMR    6
#define E_EMB  16
#define MXD    32
#define MYD    24
#define NPIX   768   /* MXD*MYD */

#define USE_ASYNC_LDS 1   /* gfx1250 async global->LDS staging */

union Frag16 { v16h v; v8h h[2]; };
union Pack8h { v8h v; _Float16 e[8]; };

// ---------------------------------------------------------------------------
// Async global->LDS b128 copy (CDNA5 ASYNCcnt path). Falls back to a VGPR
// round-trip copy when disabled.
// ---------------------------------------------------------------------------
__device__ __forceinline__ void stage_b128(_Float16* lds_dst, const _Float16* gsrc)
{
#if USE_ASYNC_LDS
    // Low 32 bits of a generic pointer to LDS == wave-relative LDS byte offset.
    unsigned lofs = (unsigned)(unsigned long long)lds_dst;
    asm volatile("global_load_async_to_lds_b128 %0, %1, off"
                 :: "v"(lofs), "v"(gsrc) : "memory");
#else
    *(uint4*)lds_dst = *(const uint4*)gsrc;
#endif
}

__device__ __forceinline__ void stage_wait()
{
#if USE_ASYNC_LDS
    asm volatile("s_wait_asynccnt 0" ::: "memory");
#endif
}

// ---------------------------------------------------------------------------
// Prep: reorder conv weights (OIHW f32) -> cout-major / K-minor f16.
// K = (ky*KSZ + kx)*CINP + cin, CINP = cin padded to a multiple of 32.
// ---------------------------------------------------------------------------
__global__ void reorder_conv_w(const float* __restrict__ w, _Float16* __restrict__ Wr,
                               int COUT, int CINR, int CINP, int KSZ)
{
    const int KTOT  = CINP * KSZ * KSZ;
    const int i     = blockIdx.x * blockDim.x + threadIdx.x;
    if (i >= COUT * KTOT) return;
    const int o   = i / KTOT;
    const int k   = i % KTOT;
    const int cin = k % CINP;
    const int kyx = k / CINP;
    const int ky  = kyx / KSZ;
    const int kx  = kyx % KSZ;
    float v = 0.0f;
    if (cin < CINR) v = w[((o * CINR + cin) * KSZ + ky) * KSZ + kx];
    Wr[i] = (_Float16)v;
}

__global__ void f2h_kernel(const float* __restrict__ a, _Float16* __restrict__ b, int n)
{
    const int i = blockIdx.x * blockDim.x + threadIdx.x;
    if (i < n) b[i] = (_Float16)a[i];
}

// ---------------------------------------------------------------------------
// Scatter rooms onto per-sample map. One block = one sample.
// Output X0 is channel-minor: [n][pix][32] f16.
// ---------------------------------------------------------------------------
__global__ void scatter_kernel(const int* __restrict__ rpos,
                               const float* __restrict__ rooms,
                               const float* __restrict__ emb,
                               _Float16* __restrict__ X0)
{
    extern __shared__ float sm[];                    // 26 * 768 floats, [ch][pix]
    const int n   = blockIdx.x;
    const int tid = threadIdx.x;

    for (int i = tid; i < 26 * NPIX; i += 256)
        sm[i] = ((i / NPIX) == 9) ? 1.0f : 0.0f;     // bg channel is all-ones
    __syncthreads();

    for (int it = tid; it < R_ROOM * WMR * HMR; it += 256) {
        const int r  = it / (WMR * HMR);
        const int wh = it % (WMR * HMR);
        const int wm = wh / HMR, hm = wh % HMR;
        const int px = rpos[(n * R_ROOM + r) * 2 + 0];
        const int py = rpos[(n * R_ROOM + r) * 2 + 1];
        const int lin = (px + wm) * MYD + (py + hm);
        const float mask = rooms[(r * 9 + 0) * (WMR * HMR) + wh];
        #pragma unroll
        for (int ch = 0; ch < 9; ++ch) {
            const float v = rooms[(r * 9 + ch) * (WMR * HMR) + wh];
            if (v != 0.0f) atomicAdd(&sm[ch * NPIX + lin], v);   // ds_add_f32
        }
        if (mask != 0.0f) {
            #pragma unroll
            for (int e = 0; e < E_EMB; ++e)
                atomicAdd(&sm[(10 + e) * NPIX + lin], emb[r * E_EMB + e] * mask);
        }
    }
    __syncthreads();

    _Float16* xo = X0 + (size_t)n * NPIX * 32;       // [pix][ch]
    for (int i = tid; i < NPIX * 32; i += 256) {
        const int p  = i >> 5;
        const int ch = i & 31;
        const float v = (ch < 26) ? sm[ch * NPIX + p] : 0.0f;
        xo[i] = (_Float16)v;
    }
}

// ---------------------------------------------------------------------------
// Implicit-GEMM conv + bias + ReLU on v_wmma_f32_16x16x32_f16.
// A = weights (M = cout), B = im2col pixels (N = pixel), K = CIN*KSZ*KSZ.
// Channel-minor maps: A-frag = 2x global b128, B-frag = 2x ds b128,
// D tile = 1x global b128 store. One block = one sample.
// Each wave owns a fixed cout tile (8 % MT == 0) -> bias/weights hoisted.
// ---------------------------------------------------------------------------
template<int CIN, int COUT, int KSZ>
__global__ void conv_wmma(const _Float16* __restrict__ Xin,   // [n][pix][CIN]
                          const _Float16* __restrict__ Wr,    // [cout][KTOT]
                          const float*    __restrict__ bias,
                          _Float16*       __restrict__ Xout)  // [n][pix][COUT]
{
    extern __shared__ _Float16 sx[];                 // [pix][CIN]
    const int n    = blockIdx.x;
    const int tid  = threadIdx.x;
    const int lane = tid & 31;
    const int wave = tid >> 5;
    const int row  = lane & 15;
    const int hi   = (lane >> 4) & 1;

    constexpr int KTOT = CIN * KSZ * KSZ;
    constexpr int MT   = COUT / 16;                  // cout tiles (A rows)
    constexpr int PT   = NPIX / 16;                  // pixel tiles (B cols)
    constexpr int PAD  = KSZ / 2;

    // Per-wave fixed cout tile
    const int mt  = wave % MT;
    const int coA = mt * 16 + row;
    const _Float16* wbase = Wr + (size_t)coA * KTOT;
    __builtin_prefetch(wbase, 0, 0);                 // global_prefetch_b8

    // Stage input sample into LDS (async DMA path on CDNA5)
    {
        const _Float16* src = Xin + (size_t)n * NPIX * CIN;
        for (int i = tid; i < NPIX * CIN / 8; i += 256)
            stage_b128(sx + i * 8, src + i * 8);
        stage_wait();
    }
    __syncthreads();

    // Hoisted bias for this wave's 8 couts
    const float4* bp = (const float4*)(bias + mt * 16 + (hi << 3));
    const float4 bb0 = bp[0], bb1 = bp[1];
    const float be[8] = { bb0.x, bb0.y, bb0.z, bb0.w, bb1.x, bb1.y, bb1.z, bb1.w };

    _Float16* xo = Xout + (size_t)n * NPIX * COUT;

    for (int pt = wave / MT; pt < PT; pt += 8 / MT) {
        const int pix = pt * 16 + row;               // B col / D col (pixel)
        const int x = pix / MYD, y = pix % MYD;

        v8f acc = {};
        auto kstep = [&](int kk) {
            const int kyx = kk / CIN;
            const int ky = kyx / KSZ, kx = kyx % KSZ;
            const int ix = x + ky - PAD, iy = y + kx - PAD;
            const bool inb = ((unsigned)ix < (unsigned)MXD) && ((unsigned)iy < (unsigned)MYD);
            const int ipix = inb ? (ix * MYD + iy) : 0;

            Frag16 a, b;
            a.h[0] = *(const v8h*)(wbase + kk + (hi << 3));
            a.h[1] = *(const v8h*)(wbase + kk + 16 + (hi << 3));

            const _Float16* bsrc = sx + (size_t)ipix * CIN + (kk % CIN) + (hi << 3);
            const v8h z = {};
            v8h b0 = *(const v8h*)(bsrc);
            v8h b1 = *(const v8h*)(bsrc + 16);
            b.h[0] = inb ? b0 : z;
            b.h[1] = inb ? b1 : z;

            acc = __builtin_amdgcn_wmma_f32_16x16x32_f16(
                      false, a.v, false, b.v, (short)0, acc, false, false);
        };
        if constexpr (KTOT / 32 <= 9) {
            #pragma unroll
            for (int kk = 0; kk < KTOT; kk += 32) kstep(kk);
        } else if constexpr (KTOT / 32 == 18) {
            #pragma unroll 6
            for (int kk = 0; kk < KTOT; kk += 32) kstep(kk);
        } else {
            #pragma unroll 5
            for (int kk = 0; kk < KTOT; kk += 32) kstep(kk);
        }

        // Epilogue: bias + ReLU, pack 8 couts -> one b128 store
        Pack8h st;
        #pragma unroll
        for (int v = 0; v < 8; ++v) {
            float val = acc[v] + be[v];
            st.e[v] = (_Float16)fmaxf(val, 0.0f);
        }
        *(v8h*)(xo + (size_t)pix * COUT + mt * 16 + (hi << 3)) = st.v;
    }
}

// ---------------------------------------------------------------------------
// Per-sample room MLP layer on WMMA. A = weights [o][64] (LDS, hoisted per
// wave), B = Y [room][ch] (LDS). M=64 out, N=64 rooms, K=64.
// ---------------------------------------------------------------------------
template<bool RELU, bool TOGLOBAL>
__device__ __forceinline__ void mlp_layer(const _Float16* __restrict__ W,  // LDS [o][64]
                                          const float*    __restrict__ bias,
                                          const _Float16* Yin,             // LDS [room][64]
                                          _Float16* Yout,                  // LDS [room][64]
                                          float* __restrict__ gout,        // out[n][room][64]
                                          int lane, int wave)
{
    const int row = lane & 15;
    const int hi  = (lane >> 4) & 1;
    const int mt  = wave & 3;                        // fixed out-channel tile
    const int oA  = mt * 16 + row;

    // Hoisted A-fragments (both K steps) and bias
    Frag16 a0, a1;
    a0.h[0] = *(const v8h*)(W + oA * 64 +  0 + (hi << 3));
    a0.h[1] = *(const v8h*)(W + oA * 64 + 16 + (hi << 3));
    a1.h[0] = *(const v8h*)(W + oA * 64 + 32 + (hi << 3));
    a1.h[1] = *(const v8h*)(W + oA * 64 + 48 + (hi << 3));
    const float4* bp = (const float4*)(bias + mt * 16 + (hi << 3));
    const float4 bb0 = bp[0], bb1 = bp[1];
    const float be[8] = { bb0.x, bb0.y, bb0.z, bb0.w, bb1.x, bb1.y, bb1.z, bb1.w };

    #pragma unroll
    for (int nt = wave >> 2; nt < 4; nt += 2) {
        const int room = nt * 16 + row;
        Frag16 b0, b1;
        b0.h[0] = *(const v8h*)(Yin + room * 64 +  0 + (hi << 3));
        b0.h[1] = *(const v8h*)(Yin + room * 64 + 16 + (hi << 3));
        b1.h[0] = *(const v8h*)(Yin + room * 64 + 32 + (hi << 3));
        b1.h[1] = *(const v8h*)(Yin + room * 64 + 48 + (hi << 3));
        v8f acc = {};
        acc = __builtin_amdgcn_wmma_f32_16x16x32_f16(
                  false, a0.v, false, b0.v, (short)0, acc, false, false);
        acc = __builtin_amdgcn_wmma_f32_16x16x32_f16(
                  false, a1.v, false, b1.v, (short)0, acc, false, false);
        if (TOGLOBAL) {
            float ov[8];
            #pragma unroll
            for (int v = 0; v < 8; ++v) {
                float val = acc[v] + be[v];
                if (RELU) val = fmaxf(val, 0.0f);
                ov[v] = val;
            }
            float4* gp = (float4*)(gout + room * 64 + mt * 16 + (hi << 3));
            gp[0] = make_float4(ov[0], ov[1], ov[2], ov[3]);
            gp[1] = make_float4(ov[4], ov[5], ov[6], ov[7]);
        } else {
            Pack8h st;
            #pragma unroll
            for (int v = 0; v < 8; ++v) {
                float val = acc[v] + be[v];
                if (RELU) val = fmaxf(val, 0.0f);
                st.e[v] = (_Float16)val;
            }
            *(v8h*)(Yout + room * 64 + mt * 16 + (hi << 3)) = st.v;
        }
    }
}

// ---------------------------------------------------------------------------
// Gather room windows (mask-weighted reduce) + 3-layer room MLP.
// One block = one sample. X2 is channel-minor [pix][64].
// ---------------------------------------------------------------------------
__global__ void gather_mlp(const _Float16* __restrict__ X2,   // [n][pix][64]
                           const float* __restrict__ rooms,
                           const int*   __restrict__ rpos,
                           const _Float16* __restrict__ rw0h, const float* __restrict__ rb0,
                           const _Float16* __restrict__ rw1h, const float* __restrict__ rb1,
                           const _Float16* __restrict__ rw2h, const float* __restrict__ rb2,
                           float* __restrict__ out)
{
    __shared__ _Float16 sw0[64 * 64], sw1[64 * 64], sw2[64 * 64];  // weights [o][c]
    __shared__ _Float16 Y0[64 * 64], Y1[64 * 64];                  // [room][ch]
    const int n    = blockIdx.x;
    const int tid  = threadIdx.x;
    const int lane = tid & 31;
    const int wave = tid >> 5;

    // Stage MLP weights into LDS
    {
        const uint4* s0 = (const uint4*)rw0h;  uint4* d0 = (uint4*)sw0;
        const uint4* s1 = (const uint4*)rw1h;  uint4* d1 = (uint4*)sw1;
        const uint4* s2 = (const uint4*)rw2h;  uint4* d2 = (uint4*)sw2;
        for (int i = tid; i < 512; i += 256) { d0[i] = s0[i]; d1[i] = s1[i]; d2[i] = s2[i]; }
    }

    // Gather: thread = (room, 16-channel group)
    {
        const _Float16* xb = X2 + (size_t)n * NPIX * 64;
        const int r  = tid >> 2;
        const int cg = (tid & 3) << 4;               // channel group base
        const int px = rpos[(n * R_ROOM + r) * 2 + 0];
        const int py = rpos[(n * R_ROOM + r) * 2 + 1];
        float acc[16];
        #pragma unroll
        for (int i = 0; i < 16; ++i) acc[i] = 0.0f;
        for (int wh = 0; wh < WMR * HMR; ++wh) {
            const int wm = wh / HMR, hm = wh % HMR;
            const float mk = rooms[(r * 9 + 0) * (WMR * HMR) + wh];
            if (mk != 0.0f) {
                const _Float16* src = xb + (size_t)((px + wm) * MYD + (py + hm)) * 64 + cg;
                Pack8h c0, c1;
                c0.v = *(const v8h*)(src);
                c1.v = *(const v8h*)(src + 8);
                #pragma unroll
                for (int i = 0; i < 8; ++i) {
                    acc[i]     += mk * (float)c0.e[i];
                    acc[i + 8] += mk * (float)c1.e[i];
                }
            }
        }
        Pack8h o0, o1;
        #pragma unroll
        for (int i = 0; i < 8; ++i) { o0.e[i] = (_Float16)acc[i]; o1.e[i] = (_Float16)acc[i + 8]; }
        *(v8h*)(Y0 + r * 64 + cg)     = o0.v;
        *(v8h*)(Y0 + r * 64 + cg + 8) = o1.v;
    }
    __syncthreads();

    mlp_layer<true,  false>(sw0, rb0, Y0, Y1, nullptr, lane, wave);
    __syncthreads();
    mlp_layer<true,  false>(sw1, rb1, Y1, Y0, nullptr, lane, wave);
    __syncthreads();
    mlp_layer<false, true >(sw2, rb2, Y0, nullptr, out + (size_t)n * R_ROOM * 64, lane, wave);
}

// ---------------------------------------------------------------------------
// Host launcher
// ---------------------------------------------------------------------------
extern "C" void kernel_launch(void* const* d_in, const int* in_sizes, int n_in,
                              void* d_out, int out_size, void* d_ws, size_t ws_size,
                              hipStream_t stream)
{
    (void)in_sizes; (void)n_in; (void)out_size; (void)ws_size;

    const int*   rpos  = (const int*)  d_in[0];
    const float* rooms = (const float*)d_in[1];
    const float* emb   = (const float*)d_in[2];
    const float* w0    = (const float*)d_in[3];
    const float* b0    = (const float*)d_in[4];
    const float* w1    = (const float*)d_in[5];
    const float* b1    = (const float*)d_in[6];
    const float* w2    = (const float*)d_in[7];
    const float* b2    = (const float*)d_in[8];
    const float* rw0   = (const float*)d_in[9];
    const float* rb0   = (const float*)d_in[10];
    const float* rw1   = (const float*)d_in[11];
    const float* rb1   = (const float*)d_in[12];
    const float* rw2   = (const float*)d_in[13];
    const float* rb2   = (const float*)d_in[14];

    char* ws = (char*)d_ws;
    size_t off = 0;
    auto carve = [&](size_t bytes) -> char* {
        char* p = ws + off;
        off = (off + bytes + 255) & ~(size_t)255;
        return p;
    };

    const size_t BUFB = (size_t)N_SAMP * NPIX * 64 * 2;      // 48 MB each
    _Float16* bufA = (_Float16*)carve(BUFB);
    _Float16* bufB = (_Float16*)carve(BUFB);
    _Float16* Wr0  = (_Float16*)carve((size_t)32 * 800 * 2);  // [cout][K]
    _Float16* Wr1  = (_Float16*)carve((size_t)64 * 288 * 2);
    _Float16* Wr2  = (_Float16*)carve((size_t)64 * 576 * 2);
    _Float16* rwh0 = (_Float16*)carve(4096 * 2);
    _Float16* rwh1 = (_Float16*)carve(4096 * 2);
    _Float16* rwh2 = (_Float16*)carve(4096 * 2);

    // Weight prep
    {
        const int t0 = 32 * 800;
        const int t1 = 64 * 288;
        const int t2 = 64 * 576;
        reorder_conv_w<<<(t0 + 255) / 256, 256, 0, stream>>>(w0, Wr0, 32, 26, 32, 5);
        reorder_conv_w<<<(t1 + 255) / 256, 256, 0, stream>>>(w1, Wr1, 64, 32, 32, 3);
        reorder_conv_w<<<(t2 + 255) / 256, 256, 0, stream>>>(w2, Wr2, 64, 64, 64, 3);
        f2h_kernel<<<16, 256, 0, stream>>>(rw0, rwh0, 4096);
        f2h_kernel<<<16, 256, 0, stream>>>(rw1, rwh1, 4096);
        f2h_kernel<<<16, 256, 0, stream>>>(rw2, rwh2, 4096);
    }

    // Scatter rooms -> X0 (f16, channel-minor, padded to 32 ch)
    scatter_kernel<<<N_SAMP, 256, 26 * NPIX * sizeof(float), stream>>>(rpos, rooms, emb, bufA);

    // Conv stack (implicit GEMM on WMMA), ping-pong buffers
    conv_wmma<32, 32, 5><<<N_SAMP, 256, (size_t)NPIX * 32 * 2, stream>>>(bufA, Wr0, b0, bufB);
    conv_wmma<32, 64, 3><<<N_SAMP, 256, (size_t)NPIX * 32 * 2, stream>>>(bufB, Wr1, b1, bufA);
    conv_wmma<64, 64, 3><<<N_SAMP, 256, (size_t)NPIX * 64 * 2, stream>>>(bufA, Wr2, b2, bufB);

    // Gather + room MLP (WMMA) -> d_out (N, R, 64) f32
    gather_mlp<<<N_SAMP, 256, 0, stream>>>(bufB, rooms, rpos,
                                           rwh0, rb0, rwh1, rb1, rwh2, rb2,
                                           (float*)d_out);
}
